// GCN_83751862272702
// MI455X (gfx1250) — matrix operand order repaired and checked by reading, not compiled, not run on hardware
//
#include <hip/hip_runtime.h>
#include <hip/hip_bf16.h>

// ---------------------------------------------------------------------------
// Types for CDNA5 WMMA (wave32): v_wmma_f32_16x16x32_bf16
// ---------------------------------------------------------------------------
typedef __bf16 bf16;
typedef __attribute__((ext_vector_type(16))) __bf16 v16bf;
typedef __attribute__((ext_vector_type(8)))  float  v8f;
typedef unsigned int u32x4 __attribute__((ext_vector_type(4)));
typedef int          i32x8 __attribute__((ext_vector_type(8)));
typedef int          i32x4 __attribute__((ext_vector_type(4)));

struct alignas(16) BF8 { bf16 v[8]; };

__device__ __forceinline__ void atomAddF(float* p, float v) {
    // relaxed, device scope -> global_atomic_add_f32 on gfx1250
    __hip_atomic_fetch_add(p, v, __ATOMIC_RELAXED, __HIP_MEMORY_SCOPE_AGENT);
}

// ---------------------------------------------------------------------------
// Elementwise helpers
// ---------------------------------------------------------------------------
__global__ void k_fill(float* __restrict__ p, float v, long long n) {
    long long i = (long long)blockIdx.x * blockDim.x + threadIdx.x;
    if (i < n) p[i] = v;
}

__global__ void k_deg(const int* __restrict__ dst, float* __restrict__ deg, int E) {
    int e = blockIdx.x * blockDim.x + threadIdx.x;
    if (e < E) atomAddF(&deg[dst[e]], 1.0f);
}

__global__ void k_dinv(float* __restrict__ deg, int n) {
    int i = blockIdx.x * blockDim.x + threadIdx.x;
    if (i < n) deg[i] = rsqrtf(deg[i]);   // deg >= 1 (self loop included)
}

__global__ void k_cvt_bf16(const float* __restrict__ in, bf16* __restrict__ out, long long n) {
    long long i = (long long)blockIdx.x * blockDim.x + threadIdx.x;
    if (i < n) out[i] = (bf16)in[i];
}

// W [K x M] f32 row-major  ->  Wt [M x K] bf16 (transposed: each output
// column's K-slice is contiguous; rows of Wt are the TDM tile rows)
__global__ void k_wt(const float* __restrict__ W, bf16* __restrict__ Wt, int K, int M) {
    long long i = (long long)blockIdx.x * blockDim.x + threadIdx.x;
    if (i < (long long)K * M) {
        int k = (int)(i / M);
        int m = (int)(i % M);
        Wt[(long long)m * K + k] = (bf16)W[i];
    }
}

// ---------------------------------------------------------------------------
// WMMA GEMM:  Cf[N x M] = A[N x K](bf16) @ Wt[M x K]^T(bf16)  (+bias)(+relu)
// Block: 256 threads = 8 waves; block tile 32(rows) x 64(cols).
// The 64-column weight slab (64 x K bf16, <=32KB) is DMA'd into LDS once per
// block by the Tensor Data Mover; B fragments then come from LDS (ds_load).
// ---------------------------------------------------------------------------
template <int K>
__global__ void __launch_bounds__(256) k_gemm(
        const bf16* __restrict__ A, const bf16* __restrict__ Wt,
        float* __restrict__ Cf, bf16* __restrict__ Cbf,
        const float* __restrict__ bias, int N, int M, int relu)
{
    __shared__ alignas(16) bf16 Bs[64 * K];   // only LDS object -> offset 0

    const int tid  = threadIdx.x;
    const int lane = tid & 31;
    const int w    = tid >> 5;                 // wave id 0..7
    const int rb0  = blockIdx.x * 32;
    const int cb0  = blockIdx.y * 64;

    // ---- TDM: Wt[cb0 .. cb0+63][0..K) -> LDS (wave 0 issues; EXEC ignored)
    if (w == 0) {
        const unsigned long long ga =
            (unsigned long long)(uintptr_t)(Wt + (size_t)cb0 * K);  // byte addr
        int remCols = M - cb0; if (remCols > 64) remCols = 64;       // OOB rows -> 0

        u32x4 g0;
        g0.x = 1u;                                         // count=1, user desc
        g0.y = 0u;                                         // lds_addr = 0 (Bs)
        g0.z = (unsigned)(ga & 0xffffffffu);               // global_addr[31:0]
        g0.w = (unsigned)((ga >> 32) & 0x1ffffffu)         // global_addr[56:32]
             | (2u << 30);                                 // type = 2 ("image")
        i32x8 g1;
        g1[0] = (1 << 16);                                 // data_size=1 (2B), mask=0
        g1[1] = (K & 0xffff) << 16;                        // tensor_dim0[15:0]
        g1[2] = ((K >> 16) & 0xffff)                       // tensor_dim0[31:16]
              | ((remCols & 0xffff) << 16);                // tensor_dim1[15:0]
        g1[3] = ((unsigned)remCols >> 16)                  // tensor_dim1[31:16]
              | ((K & 0xffff) << 16);                      // tile_dim0 = K
        g1[4] = 64;                                        // tile_dim1=64, tile_dim2=0
        g1[5] = K;                                         // tensor_dim0_stride[31:0]
        g1[6] = 0;                                         // stride0 hi | stride1 lo
        g1[7] = 0;                                         // stride1 hi
        i32x4 g2 = {0, 0, 0, 0};                           // dims 2/3 unused (2D tile)
        i32x4 g3 = {0, 0, 0, 0};
        i32x8 g4 = {0, 0, 0, 0, 0, 0, 0, 0};               // unused trailing group
        __builtin_amdgcn_tensor_load_to_lds(g0, g1, g2, g3, g4, 0);
        __builtin_amdgcn_s_wait_tensorcnt(0);
    }
    __syncthreads();

    const int r0 = rb0 + (w >> 2) * 16;        // wave row-tile base
    const int c0 = cb0 + (w & 3) * 16;         // wave col-tile base
    if (r0 >= N || c0 >= M) return;            // wave-uniform skip (after barrier)

    const int lm = lane & 15;
    const int hi = lane >> 4;                  // half-wave select
    int arow = r0 + lm; if (arow >= N) arow = N - 1;   // clamp: no EXEC churn;
                                               // OOB rows never stored below
    const bf16* __restrict__ Ap = A + (size_t)arow * K;
    const bf16* Bp = &Bs[((w & 3) * 16 + lm) * K];     // LDS, column = c0+lm

    v8f acc = {};
    #pragma unroll
    for (int k0 = 0; k0 < K; k0 += 32) {
        // A fragment: lanes 0-15 hold K {0..7,16..23}, lanes 16-31 {8..15,24..31}
        const BF8 a0 = *reinterpret_cast<const BF8*>(Ap + k0 + hi * 8);
        const BF8 a1 = *reinterpret_cast<const BF8*>(Ap + k0 + 16 + hi * 8);
        // B fragment: lanes 0-15 hold K {0..7,16..23}, lanes 16-31 {8..15,24..31}
        const BF8 b0 = *reinterpret_cast<const BF8*>(Bp + k0 + hi * 8);
        const BF8 b1 = *reinterpret_cast<const BF8*>(Bp + k0 + 16 + hi * 8);
        v16bf a, b;
        #pragma unroll
        for (int i = 0; i < 8; ++i) {
            a[i] = a0.v[i]; a[i + 8] = a1.v[i];
            b[i] = b0.v[i]; b[i + 8] = b1.v[i];
        }
        acc = __builtin_amdgcn_wmma_f32_16x16x32_bf16(
                /*neg_a=*/false, a, /*neg_b=*/false, b,
                /*c_mod=*/(short)0, acc, /*reuse_a=*/false, /*reuse_b=*/false);
    }

    const int   bcol = c0 + lm;
    const float bval = bias ? bias[bcol] : 0.0f;
    #pragma unroll
    for (int i = 0; i < 8; ++i) {              // VGPR i: rows r0+i / r0+8+i
        const int row = r0 + i + hi * 8;
        if (row < N) {
            float v = acc[i] + bval;
            if (relu) v = fmaxf(v, 0.0f);
            Cf[(size_t)row * M + bcol] = v;
            if (Cbf) Cbf[(size_t)row * M + bcol] = (bf16)v;
        }
    }
}

// ---------------------------------------------------------------------------
// Graph aggregation
// ---------------------------------------------------------------------------
// agg[i,:] = h[i,:] * dinv[i]^2   (self-loop term; also initializes agg)
__global__ void k_selfloop(const float* __restrict__ H, const float* __restrict__ dinv,
                           float* __restrict__ AGG, int N, int C) {
    long long i = (long long)blockIdx.x * blockDim.x + threadIdx.x;
    if (i < (long long)N * C) {
        int node = (int)(i / C);
        float di = dinv[node];
        AGG[i] = H[i] * di * di;
    }
}

// agg[dst,:] += h[src,:] * dinv[src]*dinv[dst] for every edge; 4 feats/thread
__global__ void k_scatter(const int* __restrict__ srcs, const int* __restrict__ dsts,
                          const float* __restrict__ dinv, const float* __restrict__ H,
                          float* __restrict__ AGG, int E, int C) {
    const int chunks = C >> 2;
    long long gid = (long long)blockIdx.x * blockDim.x + threadIdx.x;
    if (gid >= (long long)E * chunks) return;
    int e  = (int)(gid / chunks);
    int cc = (int)(gid % chunks);
    int c0 = cc << 2;
    int s = srcs[e];
    int d = dsts[e];
    if (cc == 0 && e + 8 < E) {
        // warm L2/WGP$ for an upcoming source row (global_prefetch_b8)
        __builtin_prefetch(H + (long long)srcs[e + 8] * C, 0, 1);
    }
    float nrm = dinv[s] * dinv[d];
    const float4 hv = *reinterpret_cast<const float4*>(H + (long long)s * C + c0);
    float* ap = AGG + (long long)d * C + c0;
    atomAddF(ap + 0, hv.x * nrm);
    atomAddF(ap + 1, hv.y * nrm);
    atomAddF(ap + 2, hv.z * nrm);
    atomAddF(ap + 3, hv.w * nrm);
}

// h = relu(agg + b); write f32 (for pooling) and bf16 (next layer WMMA input)
__global__ void k_bias_relu_cvt(const float* __restrict__ AGG, const float* __restrict__ b,
                                float* __restrict__ Hf, bf16* __restrict__ Hbf,
                                int N, int C) {
    long long i = (long long)blockIdx.x * blockDim.x + threadIdx.x;
    if (i < (long long)N * C) {
        int c = (int)(i % C);
        float v = fmaxf(AGG[i] + b[c], 0.0f);
        Hf[i]  = v;
        Hbf[i] = (bf16)v;
    }
}

// ---------------------------------------------------------------------------
// Global mean pool
// ---------------------------------------------------------------------------
__global__ void k_pool(const float* __restrict__ H, const int* __restrict__ batch,
                       float* __restrict__ sums, int N, int C) {
    long long i = (long long)blockIdx.x * blockDim.x + threadIdx.x;
    if (i < (long long)N * C) {
        int node = (int)(i / C);
        int c    = (int)(i % C);
        atomAddF(&sums[(long long)batch[node] * C + c], H[i]);
    }
}

__global__ void k_cnt(const int* __restrict__ batch, float* __restrict__ cnt, int N) {
    int i = blockIdx.x * blockDim.x + threadIdx.x;
    if (i < N) atomAddF(&cnt[batch[i]], 1.0f);
}

__global__ void k_hg(const float* __restrict__ sums, const float* __restrict__ cnt,
                     bf16* __restrict__ hgbf, int G, int C) {
    long long i = (long long)blockIdx.x * blockDim.x + threadIdx.x;
    if (i < (long long)G * C) {
        int g = (int)(i / C);
        hgbf[i] = (bf16)(sums[i] / fmaxf(cnt[g], 1.0f));
    }
}

// ---------------------------------------------------------------------------
// Host-side orchestration
// ---------------------------------------------------------------------------
static inline unsigned cdiv(long long a, long long b) { return (unsigned)((a + b - 1) / b); }

extern "C" void kernel_launch(void* const* d_in, const int* in_sizes, int n_in,
                              void* d_out, int out_size, void* d_ws, size_t ws_size,
                              hipStream_t stream) {
    const int F = 128, C = 256, G = 2048, DOUT = 32;
    const int N = in_sizes[0] / F;        // 50000
    const int E = in_sizes[1] / 2;        // 300000

    const float* x     = (const float*)d_in[0];
    const int*   ei    = (const int*)  d_in[1];    // [2*E]: row0 = src, row1 = dst
    const int*   batch = (const int*)  d_in[2];
    const float* W1  = (const float*)d_in[3];  const float* b1  = (const float*)d_in[4];
    const float* W2  = (const float*)d_in[5];  const float* b2  = (const float*)d_in[6];
    const float* W3  = (const float*)d_in[7];  const float* b3  = (const float*)d_in[8];
    const float* Wf1 = (const float*)d_in[9];  const float* bf1 = (const float*)d_in[10];
    const float* Wf2 = (const float*)d_in[11]; const float* bf2 = (const float*)d_in[12];
    const int* esrc = ei;
    const int* edst = ei + E;

    // Carve workspace
    char* p = (char*)d_ws;
    auto carve = [&](size_t bytes) { char* q = p; p += (bytes + 255) & ~(size_t)255; return q; };
    float* H    = (float*)carve((size_t)N * C * 4);   // GEMM out / post-relu f32
    float* AGG  = (float*)carve((size_t)N * C * 4);   // aggregation buffer
    bf16*  ABF  = (bf16*) carve((size_t)N * C * 2);   // bf16 activations (WMMA A)
    bf16*  WT   = (bf16*) carve((size_t)C * C * 2);   // transposed bf16 weights
    float* DINV = (float*)carve((size_t)N * 4);       // deg -> dinv (in place)
    float* SUMS = (float*)carve((size_t)G * C * 4);
    float* CNT  = (float*)carve((size_t)G * 4);
    bf16*  HGBF = (bf16*) carve((size_t)G * C * 2);
    float* D1F  = (float*)carve((size_t)G * F * 4);
    bf16*  D1BF = (bf16*) carve((size_t)G * F * 2);
    (void)ws_size; (void)n_in; (void)out_size;

    const int T = 256;

    // ---- degree / normalization: deg = 1 (self loop) + in-degree; dinv = deg^-1/2
    k_fill<<<cdiv(N, T), T, 0, stream>>>(DINV, 1.0f, N);
    k_deg<<<cdiv(E, T), T, 0, stream>>>(edst, DINV, E);
    k_dinv<<<cdiv(N, T), T, 0, stream>>>(DINV, N);

    // ---- x -> bf16
    k_cvt_bf16<<<cdiv((long long)N * F, T), T, 0, stream>>>(x, ABF, (long long)N * F);

    // ---- 3 GCN layers
    const float* Ws[3] = { W1, W2, W3 };
    const float* bs[3] = { b1, b2, b3 };
    for (int l = 0; l < 3; ++l) {
        const int K = (l == 0) ? F : C;
        k_wt<<<cdiv((long long)K * C, T), T, 0, stream>>>(Ws[l], WT, K, C);
        dim3 g(cdiv(N, 32), cdiv(C, 64));
        if (K == 128)
            k_gemm<128><<<g, T, 0, stream>>>(ABF, WT, H, (bf16*)nullptr,
                                             (const float*)nullptr, N, C, 0);
        else
            k_gemm<256><<<g, T, 0, stream>>>(ABF, WT, H, (bf16*)nullptr,
                                             (const float*)nullptr, N, C, 0);
        k_selfloop<<<cdiv((long long)N * C, T), T, 0, stream>>>(H, DINV, AGG, N, C);
        k_scatter<<<cdiv((long long)E * (C >> 2), T), T, 0, stream>>>(esrc, edst, DINV, H, AGG, E, C);
        k_bias_relu_cvt<<<cdiv((long long)N * C, T), T, 0, stream>>>(AGG, bs[l], H, ABF, N, C);
    }

    // ---- global mean pool
    k_fill<<<cdiv((long long)G * C, T), T, 0, stream>>>(SUMS, 0.0f, (long long)G * C);
    k_fill<<<cdiv(G, T), T, 0, stream>>>(CNT, 0.0f, G);
    k_pool<<<cdiv((long long)N * C, T), T, 0, stream>>>(H, batch, SUMS, N, C);
    k_cnt<<<cdiv(N, T), T, 0, stream>>>(batch, CNT, N);
    k_hg<<<cdiv((long long)G * C, T), T, 0, stream>>>(SUMS, CNT, HGBF, G, C);

    // ---- MLP head: relu(hg @ Wf1 + bf1) @ Wf2 + bf2
    k_wt<<<cdiv((long long)C * F, T), T, 0, stream>>>(Wf1, WT, C, F);
    {
        dim3 g(cdiv(G, 32), cdiv(F, 64));
        k_gemm<256><<<g, T, 0, stream>>>(HGBF, WT, D1F, D1BF, bf1, G, F, 1);
    }
    k_wt<<<cdiv((long long)F * DOUT, T), T, 0, stream>>>(Wf2, WT, F, DOUT);
    {
        dim3 g(cdiv(G, 32), cdiv(DOUT, 64));
        k_gemm<128><<<g, T, 0, stream>>>(D1BF, WT, (float*)d_out, (bf16*)nullptr, bf2, G, DOUT, 0);
    }
}
